// MultiHeadedAttention_90847148245354
// MI455X (gfx1250) — matrix operand order repaired and checked by reading, not compiled, not run on hardware
//
#include <hip/hip_runtime.h>
#include <hip/hip_bf16.h>

// Problem constants (match reference)
#define B_  2
#define S_  2048
#define D_  1024
#define H_  16
#define DK_ 64

// Flip to 0 to fall back to plain LDS copy staging if TDM asm fails to build.
#define ATHENA_USE_TDM 1

typedef __attribute__((ext_vector_type(16))) __bf16    v16bf;
typedef __attribute__((ext_vector_type(8)))  float     v8f;
typedef __attribute__((ext_vector_type(4)))  uint32_t  u32x4;
typedef __attribute__((ext_vector_type(8)))  uint32_t  u32x8;

union ABFrag { v16bf v; uint4 q[2]; };

__device__ __forceinline__ v8f vzero8() {
  v8f z;
#pragma unroll
  for (int i = 0; i < 8; ++i) z[i] = 0.0f;
  return z;
}

// A-fragment for V_WMMA_F32_16X16X32_BF16 (16x32, M x K), from row-major LDS tile.
// ISA layout: lanes 0-15 (row M=lane): K {0..7} in v0-3, K {16..23} in v4-7;
//             lanes 16-31 (row M=lane-16): K {8..15}, K {24..31}.
__device__ __forceinline__ v16bf load_a_frag(const __bf16* sm, int row_base, int k0, int ld) {
  const int lane = threadIdx.x & 31;
  const int m    = lane & 15;
  const int sel  = lane >> 4;
  const __bf16* p = sm + (size_t)(row_base + m) * ld + k0 + sel * 8;
  ABFrag f;
  f.q[0] = *(const uint4*)(p);
  f.q[1] = *(const uint4*)(p + 16);
  return f.v;
}

// B-fragment (32x16, K x N) from LDS staged TRANSPOSED as [n][k] row-major.
// ISA layout: lanes 0-15 (col N=lane): K {0..7},{16..23} ; lanes 16-31 (col N=lane-16): K {8..15},{24..31}.
__device__ __forceinline__ v16bf load_b_frag(const __bf16* smT, int n_base, int k0, int ld) {
  const int lane = threadIdx.x & 31;
  const int n    = lane & 15;
  const int sel  = lane >> 4;
  const __bf16* p = smT + (size_t)(n_base + n) * ld + k0 + sel * 8;
  ABFrag f;
  f.q[0] = *(const uint4*)(p);
  f.q[1] = *(const uint4*)(p + 16);
  return f.v;
}

// ---------------------------------------------------------------------------
// Tensor Data Mover: DMA a 64x64 bf16 tile (row stride `stride0` elements)
// from global into LDS at byte offset `lds_off`, inserting 4 DWORDs of pad
// after every 32 DWORDs (128B row) -> LDS row pitch 144B = 72 bf16 (our ld=72).
// D# layout per CDNA5 ISA. Tracked by TENSORcnt.
// ---------------------------------------------------------------------------
__device__ __forceinline__ void tdm_load_tile64(uint32_t lds_off, uint64_t gaddr,
                                                uint32_t tdim0, uint32_t tdim1,
                                                uint32_t stride0) {
  u32x4 g0;
  g0[0] = 1u;                                                    // count=1, user D#
  g0[1] = lds_off;                                               // lds_addr (bytes)
  g0[2] = (uint32_t)gaddr;                                       // global_addr[31:0]
  g0[3] = (uint32_t)((gaddr >> 32) & 0x01ffffffu) | (2u << 30);  // [56:32] | type=2
  u32x8 g1;
  g1[0] = (1u << 16)      // data_size = 2 bytes
        | (1u << 20)      // pad_enable
        | (4u << 22)      // pad_interval: 32 DWORDs
        | (3u << 25);     // pad_amount: 4 DWORDs
  g1[1] = (tdim0 & 0xffffu) << 16;                               // tensor_dim0 lo
  g1[2] = (tdim0 >> 16) | ((tdim1 & 0xffffu) << 16);             // dim0 hi | dim1 lo
  g1[3] = (tdim1 >> 16) | (64u << 16);                           // dim1 hi | tile_dim0=64
  g1[4] = 64u;                                                   // tile_dim1=64, tile_dim2=0
  g1[5] = stride0;                                               // tensor_dim0_stride lo
  g1[6] = 0u;
  g1[7] = 0u;
  asm volatile("tensor_load_to_lds %0, %1" :: "s"(g0), "s"(g1) : "memory");
}

__device__ __forceinline__ void wait_tensorcnt0() {
#if __has_builtin(__builtin_amdgcn_s_wait_tensorcnt)
  __builtin_amdgcn_s_wait_tensorcnt(0);
#else
  asm volatile("s_wait_tensorcnt 0x0" ::: "memory");
#endif
}
__device__ __forceinline__ void wait_tensorcnt1() {
#if __has_builtin(__builtin_amdgcn_s_wait_tensorcnt)
  __builtin_amdgcn_s_wait_tensorcnt(1);
#else
  asm volatile("s_wait_tensorcnt 0x1" ::: "memory");
#endif
}

// ---------------------------------------------------------------------------
// prop (fp32 [S,S]) -> bf16 copy, done once (8 MB, L2-resident thereafter).
// ---------------------------------------------------------------------------
__global__ __launch_bounds__(256) void prop_to_bf16_kernel(const float* __restrict__ p,
                                                           __bf16* __restrict__ o) {
  size_t i = ((size_t)blockIdx.x * 256 + threadIdx.x) * 8;
  float4 a = *(const float4*)&p[i];
  float4 b = *(const float4*)&p[i + 4];
  __bf16 tmp[8] = {(__bf16)a.x, (__bf16)a.y, (__bf16)a.z, (__bf16)a.w,
                   (__bf16)b.x, (__bf16)b.y, (__bf16)b.z, (__bf16)b.w};
  *(uint4*)&o[i] = *(uint4*)tmp;
}

// ---------------------------------------------------------------------------
// V (bf16, [B,S,H*DK]) -> VT (bf16, [B,H,DK,S]) tiled transpose through LDS,
// so attention's V staging becomes contiguous rows (TDM-friendly).
// ---------------------------------------------------------------------------
__global__ __launch_bounds__(128) void v_transpose_kernel(const __bf16* __restrict__ v,
                                                          __bf16* __restrict__ vt) {
  __shared__ __bf16 t[64 * 72];
  const int blk = blockIdx.x;
  const int lt = blk & 31, h = (blk >> 5) & (H_ - 1), bb = blk >> 9;
  const int l0 = lt * 64;
  const int tid = threadIdx.x;
  const size_t ibase = (size_t)bb * S_ * D_ + (size_t)h * DK_;
#pragma unroll
  for (int i = 0; i < 4; ++i) {
    int e = tid + i * 128;
    int l = e >> 3, c8 = (e & 7) * 8;
    *(uint4*)&t[l * 72 + c8] = *(const uint4*)&v[ibase + (size_t)(l0 + l) * D_ + c8];
  }
  __syncthreads();
  const size_t obase = ((size_t)bb * H_ + h) * DK_ * S_;
#pragma unroll
  for (int i = 0; i < 4; ++i) {
    int e = tid + i * 128;
    int dk = e >> 3, l8 = (e & 7) * 8;
    __bf16 tmp[8];
#pragma unroll
    for (int j = 0; j < 8; ++j) tmp[j] = t[(l8 + j) * 72 + dk];
    *(uint4*)&vt[obase + (size_t)dk * S_ + l0 + l8] = *(uint4*)tmp;
  }
}

// ---------------------------------------------------------------------------
// Generic tiled GEMM: Y[M,N] = A[M,K] fp32 * W[K,N] fp32 + bias, bf16 WMMA,
// fp32 accumulation. 128 threads = 4 waves; tile 64x64, K-step 32.
// ---------------------------------------------------------------------------
__global__ __launch_bounds__(128) void proj_gemm_kernel(
    const float* __restrict__ A, const float* __restrict__ W,
    const float* __restrict__ bias,
    __bf16* __restrict__ Ybf, float* __restrict__ Yf,
    int M, int N, int K) {
  __shared__ __bf16 Asm[64 * 40];
  __shared__ __bf16 Bsm[64 * 40];
  const int LDA = 40, LDB = 40;
  const int tid = threadIdx.x, wave = tid >> 5, lane = tid & 31;
  const int row0 = blockIdx.y * 64, col0 = blockIdx.x * 64;

  v8f acc[4];
#pragma unroll
  for (int t = 0; t < 4; ++t) acc[t] = vzero8();

  for (int k0 = 0; k0 < K; k0 += 32) {
#pragma unroll
    for (int i = 0; i < 4; ++i) {
      int e = tid + i * 128;
      int r = e >> 3, k4 = (e & 7) * 4;
      float4 av = *(const float4*)&A[(size_t)(row0 + r) * K + k0 + k4];
      __bf16* d = &Asm[r * LDA + k4];
      d[0] = (__bf16)av.x; d[1] = (__bf16)av.y; d[2] = (__bf16)av.z; d[3] = (__bf16)av.w;
    }
#pragma unroll
    for (int i = 0; i < 4; ++i) {
      int e = tid + i * 128;
      int k = e >> 4, n4 = (e & 15) * 4;
      float4 wv = *(const float4*)&W[(size_t)(k0 + k) * N + col0 + n4];
      Bsm[(n4 + 0) * LDB + k] = (__bf16)wv.x;
      Bsm[(n4 + 1) * LDB + k] = (__bf16)wv.y;
      Bsm[(n4 + 2) * LDB + k] = (__bf16)wv.z;
      Bsm[(n4 + 3) * LDB + k] = (__bf16)wv.w;
    }
    __syncthreads();
    v16bf afr = load_a_frag(Asm, wave * 16, 0, LDA);
#pragma unroll
    for (int t = 0; t < 4; ++t) {
      v16bf bfr = load_b_frag(Bsm, t * 16, 0, LDB);
      acc[t] = __builtin_amdgcn_wmma_f32_16x16x32_bf16(
          false, afr, false, bfr, (short)0, acc[t], false, false);
    }
    __syncthreads();
  }

  const int hf = lane >> 4, cl = lane & 15;
#pragma unroll
  for (int t = 0; t < 4; ++t) {
#pragma unroll
    for (int r = 0; r < 8; ++r) {
      int rr = row0 + wave * 16 + r + hf * 8;
      int cc = col0 + t * 16 + cl;
      float v = acc[t][r] + bias[cc];
      if (Ybf) Ybf[(size_t)rr * N + cc] = (__bf16)v;
      else     Yf [(size_t)rr * N + cc] = v;
    }
  }
}

// ---------------------------------------------------------------------------
// Fused attention: per workgroup = one (b, h, 64-query tile).
// Phase A: S_row[64][2048] = mask0(Q @ K^T)/8 in LDS (bf16, ~257 KB).
// Phase B: flash loop over l-blocks: P = S_row @ prop^T, -inf mask,
//          online softmax, o += exp(P) @ V.  All tile staging via TDM DMA
//          (double-buffered, overlapped with WMMA / softmax VALU).
// ---------------------------------------------------------------------------
#define SROW_LD    2056
#define SROW_BYTES (64 * SROW_LD * 2)            // 263168
#define BUFT_BYTES (64 * 72 * 2)                 // 9216
#define SMEM_ATTN_BYTES (SROW_BYTES + 3 * BUFT_BYTES)   // 290816 <= 320K

__global__ __launch_bounds__(128) void attn_fused_kernel(
    const __bf16* __restrict__ Qbf, const __bf16* __restrict__ Kbf,
    const __bf16* __restrict__ VbfT,
    const int* __restrict__ mask, const __bf16* __restrict__ prop_bf,
    float* __restrict__ attn_out) {
  extern __shared__ char smem_raw[];
  __bf16* S_row = (__bf16*)smem_raw;                       // [64][SROW_LD]
  __bf16* bufA  = (__bf16*)(smem_raw + SROW_BYTES);        // [64][72] Q tile -> exp(P) staging
  __bf16* bufP0 = bufA + 64 * 72;                          // [64][72] stage buffer 0
  __bf16* bufP1 = bufP0 + 64 * 72;                         // [64][72] stage buffer 1
  const __bf16* bufP[2] = {bufP0, bufP1};
  const uint32_t lds0  = (uint32_t)(uintptr_t)smem_raw;    // LDS byte offset of smem base
  const uint32_t offA  = lds0 + SROW_BYTES;
  const uint32_t offP0 = offA + BUFT_BYTES;
  const uint32_t offP1 = offP0 + BUFT_BYTES;

  const int blk = blockIdx.x;
  const int qt = blk & 31;
  const int h  = (blk >> 5) & (H_ - 1);
  const int bb = blk >> 9;
  const int q0 = qt * 64;
  const int tid = threadIdx.x, wave = tid >> 5, lane = tid & 31;
  const int hf = lane >> 4, cl = lane & 15;

  const size_t hbase = (size_t)bb * S_ * D_ + (size_t)h * DK_;
  const int* mrow = mask + (size_t)bb * S_ * S_;
  const uint64_t q_gbase  = (uint64_t)(uintptr_t)(Qbf + hbase);
  const uint64_t k_gbase  = (uint64_t)(uintptr_t)(Kbf + hbase);
  const uint64_t p_gbase  = (uint64_t)(uintptr_t)prop_bf;
  const uint64_t vt_gbase = (uint64_t)(uintptr_t)(VbfT + ((size_t)bb * H_ + h) * DK_ * S_);

#if ATHENA_USE_TDM
  if (wave == 0) {
    tdm_load_tile64(offA,  q_gbase + (size_t)q0 * D_ * 2, D_, S_, D_);  // Q tile
    tdm_load_tile64(offP0, k_gbase, D_, S_, D_);                        // K tile kb=0
  }
#else
#pragma unroll
  for (int i = 0; i < 4; ++i) {
    int e = tid + i * 128;
    int r = e >> 3, c8 = (e & 7) * 8;
    *(uint4*)&bufA[r * 72 + c8] = *(const uint4*)&Qbf[hbase + (size_t)(q0 + r) * D_ + c8];
  }
  __syncthreads();
#endif

  // ---------------- Phase A: masked, scaled scores into LDS ----------------
  for (int kb = 0; kb < 32; ++kb) {
    const int kcol0 = kb * 64;
#if ATHENA_USE_TDM
    if (wave == 0) {
      if (kb + 1 < 32) {
        tdm_load_tile64(((kb + 1) & 1) ? offP1 : offP0,
                        k_gbase + (size_t)(kb + 1) * 64 * D_ * 2, D_, S_, D_);
        wait_tensorcnt1();
      } else {
        wait_tensorcnt0();
      }
    }
    const __bf16* kbuf = bufP[kb & 1];
#else
#pragma unroll
    for (int i = 0; i < 4; ++i) {
      int e = tid + i * 128;
      int r = e >> 3, c8 = (e & 7) * 8;
      *(uint4*)&bufP0[r * 72 + c8] =
          *(const uint4*)&Kbf[hbase + (size_t)(kcol0 + r) * D_ + c8];
    }
    const __bf16* kbuf = bufP0;
#endif
    __syncthreads();

    v8f acc[4];
#pragma unroll
    for (int t = 0; t < 4; ++t) acc[t] = vzero8();
#pragma unroll
    for (int ks = 0; ks < 64; ks += 32) {
      v16bf afr = load_a_frag(bufA, wave * 16, ks, 72);
#pragma unroll
      for (int t = 0; t < 4; ++t) {
        v16bf bfr = load_b_frag(kbuf, t * 16, ks, 72);
        acc[t] = __builtin_amdgcn_wmma_f32_16x16x32_bf16(
            false, afr, false, bfr, (short)0, acc[t], false, false);
      }
    }
#pragma unroll
    for (int t = 0; t < 4; ++t) {
#pragma unroll
      for (int r = 0; r < 8; ++r) {
        int rq = wave * 16 + r + hf * 8;
        int ck = kcol0 + t * 16 + cl;
        int mv = mrow[(size_t)(q0 + rq) * S_ + ck];
        float v = mv ? acc[t][r] * 0.125f : 0.0f;
        S_row[rq * SROW_LD + ck] = (__bf16)v;
      }
    }
    __syncthreads();
  }

  // ---------------- Phase B: propagate + online softmax + @V ----------------
  __bf16* p_st = bufA + wave * (16 * 72);
  float m_run[8], l_run[8];
  v8f o_acc[4];
#pragma unroll
  for (int r = 0; r < 8; ++r) { m_run[r] = -__builtin_inff(); l_run[r] = 0.0f; }
#pragma unroll
  for (int t = 0; t < 4; ++t) o_acc[t] = vzero8();

  for (int lb = 0; lb < 32; ++lb) {
    const int l0 = lb * 64;
    v8f accP[4];
#pragma unroll
    for (int t = 0; t < 4; ++t) accP[t] = vzero8();

#if ATHENA_USE_TDM
    if (wave == 0)
      tdm_load_tile64(offP0, p_gbase + ((size_t)l0 * S_) * 2, S_, S_, S_);  // prop kb=0
#endif
    for (int kb = 0; kb < 32; ++kb) {
#if ATHENA_USE_TDM
      if (wave == 0) {
        if (kb + 1 < 32) {
          tdm_load_tile64(((kb + 1) & 1) ? offP1 : offP0,
                          p_gbase + ((size_t)l0 * S_ + (size_t)(kb + 1) * 64) * 2, S_, S_, S_);
          wait_tensorcnt1();
        } else {
          wait_tensorcnt0();
        }
      }
      const __bf16* pbuf = bufP[kb & 1];
#else
#pragma unroll
      for (int i = 0; i < 4; ++i) {
        int e = tid + i * 128;
        int l = e >> 3, c8 = (e & 7) * 8;
        *(uint4*)&bufP0[l * 72 + c8] =
            *(const uint4*)&prop_bf[((size_t)(l0 + l)) * S_ + kb * 64 + c8];
      }
      const __bf16* pbuf = bufP0;
#endif
      __syncthreads();
#pragma unroll
      for (int ks = 0; ks < 64; ks += 32) {
        v16bf afr = load_a_frag(S_row, wave * 16, kb * 64 + ks, SROW_LD);
#pragma unroll
        for (int t = 0; t < 4; ++t) {
          v16bf bfr = load_b_frag(pbuf, t * 16, ks, 72);
          accP[t] = __builtin_amdgcn_wmma_f32_16x16x32_bf16(
              false, afr, false, bfr, (short)0, accP[t], false, false);
        }
      }
      __syncthreads();
    }

#if ATHENA_USE_TDM
    // Kick the V^T tile DMA now; it overlaps the softmax VALU work below.
    if (wave == 0)
      tdm_load_tile64(offP0, vt_gbase + (size_t)l0 * 2, S_, S_, S_);
#else
#pragma unroll
    for (int i = 0; i < 4; ++i) {
      int e = tid + i * 128;
      int dk = e >> 3, l8 = (e & 7) * 8;
      *(uint4*)&bufP0[dk * 72 + l8] =
          *(const uint4*)&VbfT[((size_t)bb * H_ + h) * DK_ * S_ + (size_t)dk * S_ + l0 + l8];
    }
#endif

    // -inf mask over l
#pragma unroll
    for (int t = 0; t < 4; ++t) {
#pragma unroll
      for (int r = 0; r < 8; ++r) {
        int rq = wave * 16 + r + hf * 8;
        int cc = l0 + t * 16 + cl;
        if (!mrow[(size_t)(q0 + rq) * S_ + cc]) accP[t][r] = -__builtin_inff();
      }
    }

    // Online softmax per row (rows live in 16-lane halves)
#pragma unroll
    for (int r = 0; r < 8; ++r) {
      float mx = fmaxf(fmaxf(accP[0][r], accP[1][r]), fmaxf(accP[2][r], accP[3][r]));
#pragma unroll
      for (int off = 8; off >= 1; off >>= 1) mx = fmaxf(mx, __shfl_xor(mx, off, 32));
      float newm = fmaxf(m_run[r], mx);
      float alpha = 1.0f, psum = 0.0f;
      if (newm != -__builtin_inff()) {
        alpha = __expf(m_run[r] - newm);
#pragma unroll
        for (int t = 0; t < 4; ++t) {
          float p = __expf(accP[t][r] - newm);
          psum += p;
          accP[t][r] = p;
        }
      } else {
#pragma unroll
        for (int t = 0; t < 4; ++t) accP[t][r] = 0.0f;
      }
#pragma unroll
      for (int off = 8; off >= 1; off >>= 1) psum += __shfl_xor(psum, off, 32);
      l_run[r] = l_run[r] * alpha + psum;
      m_run[r] = newm;
#pragma unroll
      for (int t = 0; t < 4; ++t) o_acc[t][r] *= alpha;
#pragma unroll
      for (int t = 0; t < 4; ++t)
        p_st[(r + hf * 8) * 72 + t * 16 + cl] = (__bf16)accP[t][r];
    }

#if ATHENA_USE_TDM
    if (wave == 0) wait_tensorcnt0();
#endif
    __syncthreads();

    // o_acc += exp(P)[16x64] @ V_blk[64x64]
#pragma unroll
    for (int ks = 0; ks < 64; ks += 32) {
      v16bf afr = load_a_frag(p_st, 0, ks, 72);
#pragma unroll
      for (int t = 0; t < 4; ++t) {
        v16bf bfr = load_b_frag(bufP0, t * 16, ks, 72);
        o_acc[t] = __builtin_amdgcn_wmma_f32_16x16x32_bf16(
            false, afr, false, bfr, (short)0, o_acc[t], false, false);
      }
    }
    __syncthreads();
  }

  // Final: divide by softmax denominator (nan_to_num for fully-masked rows)
#pragma unroll
  for (int t = 0; t < 4; ++t) {
#pragma unroll
    for (int r = 0; r < 8; ++r) {
      int rq = wave * 16 + r + hf * 8;
      float denom = l_run[r];
      float v = (denom > 0.0f) ? o_acc[t][r] / denom : 0.0f;
      attn_out[hbase + (size_t)(q0 + rq) * D_ + t * 16 + cl] = v;
    }
  }
}

// ---------------------------------------------------------------------------
extern "C" void kernel_launch(void* const* d_in, const int* in_sizes, int n_in,
                              void* d_out, int out_size, void* d_ws, size_t ws_size,
                              hipStream_t stream) {
  (void)in_sizes; (void)n_in; (void)out_size; (void)ws_size;
  const float* in_q = (const float*)d_in[0];
  const float* in_k = (const float*)d_in[1];
  const float* in_v = (const float*)d_in[2];
  const int*   mask = (const int*)  d_in[3];
  const float* prop = (const float*)d_in[4];
  const float* Wq = (const float*)d_in[5];  const float* bq = (const float*)d_in[6];
  const float* Wk = (const float*)d_in[7];  const float* bk = (const float*)d_in[8];
  const float* Wv = (const float*)d_in[9];  const float* bv = (const float*)d_in[10];
  const float* Wo = (const float*)d_in[11]; const float* bo = (const float*)d_in[12];
  float* out = (float*)d_out;

  const size_t BSD = (size_t)B_ * S_ * D_;
  const size_t SS  = (size_t)S_ * S_;
  __bf16* Qbf    = (__bf16*)d_ws;
  __bf16* Kbf    = Qbf + BSD;
  __bf16* Vbf    = Kbf + BSD;
  __bf16* VbfT   = Vbf + BSD;
  __bf16* propbf = VbfT + BSD;
  float*  attn   = (float*)(propbf + SS);

  const dim3 gblk(128);
  const dim3 ggrid(D_ / 64, (B_ * S_) / 64);

  // Q/K/V projections (fp32 in, bf16 out, fp32 WMMA accumulation)
  proj_gemm_kernel<<<ggrid, gblk, 0, stream>>>(in_q, Wq, bq, Qbf, nullptr, B_ * S_, D_, D_);
  proj_gemm_kernel<<<ggrid, gblk, 0, stream>>>(in_k, Wk, bk, Kbf, nullptr, B_ * S_, D_, D_);
  proj_gemm_kernel<<<ggrid, gblk, 0, stream>>>(in_v, Wv, bv, Vbf, nullptr, B_ * S_, D_, D_);

  // One-time bf16 prep: prop -> bf16 (L2-resident), V -> [B,H,DK,S] transpose
  prop_to_bf16_kernel<<<dim3((unsigned)(SS / 2048)), dim3(256), 0, stream>>>(prop, propbf);
  v_transpose_kernel<<<dim3(B_ * H_ * (S_ / 64)), gblk, 0, stream>>>(Vbf, VbfT);

  // Fused attention: 1024 workgroups, ~284 KB dynamic LDS (CDNA5: 320 KB/WGP)
  hipFuncSetAttribute(reinterpret_cast<const void*>(attn_fused_kernel),
                      hipFuncAttributeMaxDynamicSharedMemorySize, SMEM_ATTN_BYTES);
  attn_fused_kernel<<<dim3(B_ * H_ * (S_ / 64)), gblk, SMEM_ATTN_BYTES, stream>>>(
      Qbf, Kbf, VbfT, mask, propbf, attn);

  // Output projection (fp32 out)
  proj_gemm_kernel<<<ggrid, gblk, 0, stream>>>(attn, Wo, bo, nullptr, out, B_ * S_, D_, D_);
}
